// GraphN_11776800326051
// MI455X (gfx1250) — matrix operand order repaired and checked by reading, not compiled, not run on hardware
//
#include <hip/hip_runtime.h>
#include <hip/hip_bf16.h>
#include <math.h>

// ---------------------------------------------------------------------------
// GAT + TopKPool cascade for MI455X (gfx1250, wave32).
// GEMMs run through v_wmma_f32_16x16x32_bf16 (f32 -> bf16 RNE conversion is
// fused into the LDS staging, f32 accumulate). Graph softmax/aggregation use
// native f32 global atomics; float-max via order-preserving u32 encoding.
// ---------------------------------------------------------------------------

#define HH    3
#define CC    256
#define HC    768               // H*C
#define BGR   32
#define NPG0  512
#define DEG   8
#define N0    (BGR * NPG0)      // 16384 nodes, level 0
#define EDG   (N0 * DEG)        // 131072 edges (fixed slot count, masked later)
#define KP1   410
#define KP2   328
#define KP3   263

typedef __attribute__((ext_vector_type(16))) __bf16 v16bf;
typedef __attribute__((ext_vector_type(8)))  float  v8f;

// ---------------- helpers ----------------
__device__ inline unsigned short f2bf(float f) {
  unsigned u = __float_as_uint(f);
  unsigned r = u + 0x7FFFu + ((u >> 16) & 1u);   // round-to-nearest-even
  return (unsigned short)(r >> 16);
}
__device__ inline unsigned pack2(float lo, float hi) {
  return (unsigned)f2bf(lo) | ((unsigned)f2bf(hi) << 16);
}
__device__ inline unsigned fenc(float f) {       // order-preserving f32->u32
  unsigned u = __float_as_uint(f);
  return (u & 0x80000000u) ? ~u : (u | 0x80000000u);
}
__device__ inline float fdec(unsigned u) {
  unsigned b = (u & 0x80000000u) ? (u & 0x7FFFFFFFu) : ~u;
  return __uint_as_float(b);
}
__device__ inline bool edge_resolve(int i, const int* s, const int* d,
                                    const int* m, int E, int& si, int& di) {
  if (i < E) { si = s[i]; di = d[i]; return m ? (m[i] != 0) : true; }
  si = di = i - E;                                // self loop, always valid
  return true;
}

// ---------------- WMMA bf16 GEMM:  C[M,N] = A[M,K] @ B[K,N] (+bias, +lrelu) --
// Block: 256 thr = 8 waves; tile 128x128, BK=32; wave tile 32x64 (2x4 wmma).
#define LDT 34

__global__ __launch_bounds__(256) void k_gemm_bf16(
    const float* __restrict__ A, const float* __restrict__ Bw,
    const float* __restrict__ bias, float* __restrict__ C,
    int M, int N, int K, int act)
{
  __shared__ unsigned short lA[128 * LDT];
  __shared__ unsigned short lB[128 * LDT];
  const int tid  = threadIdx.x;
  const int lane = tid & 31;
  const int w    = tid >> 5;
  const int wm   = (w & 3) * 32;      // wave M offset in tile
  const int wn   = (w >> 2) * 64;     // wave N offset in tile
  const int bm   = blockIdx.y * 128;
  const int bn   = blockIdx.x * 128;

  v8f acc[2][4];
#pragma unroll
  for (int a = 0; a < 2; ++a)
#pragma unroll
    for (int b = 0; b < 4; ++b)
#pragma unroll
      for (int e = 0; e < 8; ++e) acc[a][b][e] = 0.0f;

  const int arow  = tid >> 1;             // 0..127
  const int ahalf = (tid & 1) * 16;       // 0 or 16
  const int bk    = tid >> 3;             // 0..31
  const int bcol  = (tid & 7) * 16;       // 0..112

  union Frag { v16bf v; unsigned u[8]; };

  for (int kt = 0; kt < K; kt += 32) {
    // ---- stage A tile (f32 -> bf16) ----
    unsigned* dstA = (unsigned*)&lA[arow * LDT + ahalf];
    if (bm + arow < M) {
      const float4* p = (const float4*)(A + (size_t)(bm + arow) * K + kt + ahalf);
      float4 a0 = p[0], a1 = p[1], a2 = p[2], a3 = p[3];
      if (kt + 32 < K)
        __builtin_prefetch(A + (size_t)(bm + arow) * K + kt + 32 + ahalf, 0, 1);
      dstA[0] = pack2(a0.x, a0.y); dstA[1] = pack2(a0.z, a0.w);
      dstA[2] = pack2(a1.x, a1.y); dstA[3] = pack2(a1.z, a1.w);
      dstA[4] = pack2(a2.x, a2.y); dstA[5] = pack2(a2.z, a2.w);
      dstA[6] = pack2(a3.x, a3.y); dstA[7] = pack2(a3.z, a3.w);
    } else {
#pragma unroll
      for (int j = 0; j < 8; ++j) dstA[j] = 0u;
    }
    // ---- stage B tile transposed: lB[col][k] ----
    {
      const float4* q = (const float4*)(Bw + (size_t)(kt + bk) * N + bn + bcol);
      float4 b0 = q[0], b1 = q[1], b2 = q[2], b3 = q[3];
      float bb[16] = { b0.x, b0.y, b0.z, b0.w, b1.x, b1.y, b1.z, b1.w,
                       b2.x, b2.y, b2.z, b2.w, b3.x, b3.y, b3.z, b3.w };
#pragma unroll
      for (int i = 0; i < 16; ++i) lB[(bcol + i) * LDT + bk] = f2bf(bb[i]);
    }
    __syncthreads();

    // ---- build fragments per ISA lane layout & issue WMMAs ----
    const int kb = (lane >> 4) * 8;       // A: lanes16-31 hold K+8 half
    Frag af[2];
#pragma unroll
    for (int tm = 0; tm < 2; ++tm) {
      const unsigned short* bp = &lA[(wm + tm * 16 + (lane & 15)) * LDT];
#pragma unroll
      for (int j = 0; j < 4; ++j) {
        af[tm].u[j]     = *(const unsigned*)&bp[kb + 2 * j];
        af[tm].u[4 + j] = *(const unsigned*)&bp[16 + kb + 2 * j];
      }
    }
    const int koffB = (lane >> 4) * 8;    // B: lanes16-31 hold K+8 half (same as A)
    Frag bfr[4];
#pragma unroll
    for (int tn = 0; tn < 4; ++tn) {
      const unsigned short* bp = &lB[(wn + tn * 16 + (lane & 15)) * LDT];
#pragma unroll
      for (int j = 0; j < 4; ++j) {
        bfr[tn].u[j]     = *(const unsigned*)&bp[koffB + 2 * j];
        bfr[tn].u[4 + j] = *(const unsigned*)&bp[16 + koffB + 2 * j];
      }
    }
#pragma unroll
    for (int tm = 0; tm < 2; ++tm)
#pragma unroll
      for (int tn = 0; tn < 4; ++tn)
        acc[tm][tn] = __builtin_amdgcn_wmma_f32_16x16x32_bf16(
            false, af[tm].v, false, bfr[tn].v, (short)0, acc[tm][tn],
            false, false);
    __syncthreads();
  }

  // ---- epilogue: C/D layout — lane&15 = N, (lane>>4)*8+v = M ----
  const int rbase = (lane >> 4) * 8;
  const int cbase = lane & 15;
#pragma unroll
  for (int tm = 0; tm < 2; ++tm)
#pragma unroll
    for (int tn = 0; tn < 4; ++tn) {
      const int col = bn + wn + tn * 16 + cbase;
      const float bv = bias ? bias[col] : 0.0f;
#pragma unroll
      for (int v = 0; v < 8; ++v) {
        const int row = bm + wm + tm * 16 + rbase + v;
        if (row < M) {
          float x = acc[tm][tn][v] + bv;
          if (act) x = x >= 0.0f ? x : 0.01f * x;
          C[(size_t)row * N + col] = x;
        }
      }
    }
}

// ---------------- misc small kernels ----------------
__global__ void k_zero_u32(unsigned* p, int n) {
  int i = blockIdx.x * blockDim.x + threadIdx.x;
  if (i < n) p[i] = 0u;
}
__global__ void k_init_out_bias(float* out, const float* b, int Nn) {
  size_t i = (size_t)blockIdx.x * blockDim.x + threadIdx.x;
  if (i < (size_t)Nn * HC) out[i] = b[i % HC];
}
__global__ void k_attn_dots(const float* __restrict__ hg,
                            const float* __restrict__ as,
                            const float* __restrict__ ad,
                            float* asrc, float* adst, int Nn) {
  int i = blockIdx.x * blockDim.x + threadIdx.x;
  if (i >= Nn * HH) return;
  int n = i / HH, h = i % HH;
  const float* hp = hg + (size_t)n * HC + h * CC;
  const float* ap = as + h * CC;
  const float* bp = ad + h * CC;
  float s1 = 0.f, s2 = 0.f;
  for (int c = 0; c < CC; ++c) { float v = hp[c]; s1 += v * ap[c]; s2 += v * bp[c]; }
  asrc[i] = s1; adst[i] = s2;
}
__global__ void k_edge_logits(const int* s, const int* d, const int* m, int E, int Nn,
                              const float* asrc, const float* adst,
                              float* ebuf, unsigned* emaxu) {
  int i = blockIdx.x * blockDim.x + threadIdx.x;
  if (i >= E + Nn) return;
  int si, di; bool ok = edge_resolve(i, s, d, m, E, si, di);
#pragma unroll
  for (int h = 0; h < HH; ++h) {
    float e = asrc[si * HH + h] + adst[di * HH + h];
    e = e >= 0.f ? e : 0.2f * e;                 // leaky_relu(0.2)
    ebuf[(size_t)i * HH + h] = e;
    if (ok) atomicMax(&emaxu[di * HH + h], fenc(e));
  }
}
__global__ void k_edge_w(const int* s, const int* d, const int* m, int E, int Nn,
                         float* ebuf, const unsigned* emaxu, float* denom) {
  int i = blockIdx.x * blockDim.x + threadIdx.x;
  if (i >= E + Nn) return;
  int si, di; bool ok = edge_resolve(i, s, d, m, E, si, di);
  (void)si;
#pragma unroll
  for (int h = 0; h < HH; ++h) {
    if (ok) {
      float wv = __expf(ebuf[(size_t)i * HH + h] - fdec(emaxu[di * HH + h]));
      ebuf[(size_t)i * HH + h] = wv;
      atomicAdd(&denom[di * HH + h], wv);
    } else {
      ebuf[(size_t)i * HH + h] = 0.f;
    }
  }
}
__global__ void k_edge_agg(const int* s, const int* d, const int* m, int E, int Nn,
                           const float* ebuf, const float* denom,
                           const float* __restrict__ hg, float* out) {
  int i = blockIdx.x;
  int si, di; bool ok = edge_resolve(i, s, d, m, E, si, di);
  if (!ok) return;
  int c = threadIdx.x;
#pragma unroll
  for (int h = 0; h < HH; ++h) {
    float alpha = ebuf[(size_t)i * HH + h] / denom[di * HH + h];
    atomicAdd(&out[(size_t)di * HC + h * CC + c],
              alpha * hg[(size_t)si * HC + h * CC + c]);
  }
}
__global__ void k_pw_invnorm(const float* pw, float* invn) {
  __shared__ float red[256];
  int t = threadIdx.x; float s = 0.f;
  for (int j = t; j < HC; j += 256) s += pw[j] * pw[j];
  red[t] = s; __syncthreads();
  for (int st = 128; st > 0; st >>= 1) {
    if (t < st) red[t] += red[t + st];
    __syncthreads();
  }
  if (t == 0) invn[0] = rsqrtf(red[0]);
}
__global__ void k_scores(const float* __restrict__ hgat, const float* pw,
                         const float* invn, float* score) {
  __shared__ float red[256];
  int n = blockIdx.x, t = threadIdx.x;
  float s = 0.f;
  for (int j = t; j < HC; j += 256) s += hgat[(size_t)n * HC + j] * pw[j];
  red[t] = s; __syncthreads();
  for (int st = 128; st > 0; st >>= 1) {
    if (t < st) red[t] += red[t + st];
    __syncthreads();
  }
  if (t == 0) score[n] = tanhf(red[0] * invn[0]);
}
__global__ void k_topk(const float* score, int n_per, int k, int* perm, int* newid) {
  __shared__ float vv[512];
  __shared__ float rv[256];
  __shared__ int   ri[256];
  int g = blockIdx.x, t = threadIdx.x;
  int base = g * n_per;
  for (int i = t; i < 512; i += 256) vv[i] = (i < n_per) ? score[base + i] : -3.4e38f;
  for (int i = t; i < n_per; i += 256) newid[base + i] = -1;
  __syncthreads();
  for (int r = 0; r < k; ++r) {
    float a = vv[t]; int ia = t;
    float b = vv[t + 256];
    if (b > a) { a = b; ia = t + 256; }           // ties -> lower index
    rv[t] = a; ri[t] = ia;
    __syncthreads();
    for (int st = 128; st > 0; st >>= 1) {
      if (t < st) {
        if (rv[t + st] > rv[t] ||
            (rv[t + st] == rv[t] && ri[t + st] < ri[t])) {
          rv[t] = rv[t + st]; ri[t] = ri[t + st];
        }
      }
      __syncthreads();
    }
    if (t == 0) {
      int best = ri[0];
      perm[g * k + r] = base + best;
      newid[base + best] = g * k + r;
      vv[best] = -3.4e38f;
    }
    __syncthreads();
  }
}
__global__ void k_gather(const float* __restrict__ hgat, const float* score,
                         const int* perm, float* xpool) {
  int i = blockIdx.x;
  int p = perm[i];
  float sc = score[p];
  for (int j = threadIdx.x; j < HC; j += 256)
    xpool[(size_t)i * HC + j] = hgat[(size_t)p * HC + j] * sc;
}
__global__ void k_remap(const int* s_in, const int* d_in, const int* m_in,
                        const int* newid, int* s_out, int* d_out, int* m_out, int E) {
  int i = blockIdx.x * blockDim.x + threadIdx.x;
  if (i >= E) return;
  int mo = m_in ? m_in[i] : 1;
  int s2 = newid[s_in[i]], d2 = newid[d_in[i]];
  int ok = (mo != 0) && (s2 >= 0) && (d2 >= 0);
  s_out[i] = ok ? s2 : 0;
  d_out[i] = ok ? d2 : 0;
  m_out[i] = ok;
}
__global__ void k_readout(const float* __restrict__ hiW, int k, float* zsum) {
  int g = blockIdx.x, c = threadIdx.x;       // 256 threads = channels
  float mx = -3.4e38f, sm = 0.f;
  const float* p = hiW + (size_t)g * k * CC + c;
  for (int r = 0; r < k; ++r) { float v = p[(size_t)r * CC]; mx = fmaxf(mx, v); sm += v; }
  zsum[g * 512 + c]       += mx;
  zsum[g * 512 + 256 + c] += sm / (float)k;
}
__global__ void k_mlp2(const float* z1, const float* l2W, const float* l2b, float* out) {
  int t = threadIdx.x;
  if (t >= 64) return;
  int g = t >> 1, o = t & 1;
  float a = l2b[o];
  for (int j = 0; j < 1024; ++j) a += z1[g * 1024 + j] * l2W[j * 2 + o];
  out[g * 2 + o] = a;
}

// ---------------- host-side sequencing ----------------
static void gemm(const float* A, const float* Bw, const float* bias, float* C,
                 int M, int N, int K, int act, hipStream_t st) {
  dim3 grid(N / 128, (M + 127) / 128);
  k_gemm_bf16<<<grid, 256, 0, st>>>(A, Bw, bias, C, M, N, K, act);
}
static void run_attention(const float* hg, const float* as, const float* ad,
                          const float* b, const int* s, const int* d, const int* m,
                          int Nn, float* agg, float* asrc, float* adst,
                          unsigned* emaxu, float* denom, float* ebuf, hipStream_t st) {
  int E2 = EDG + Nn;
  size_t tot = (size_t)Nn * HC;
  k_init_out_bias<<<(unsigned)((tot + 255) / 256), 256, 0, st>>>(agg, b, Nn);
  k_zero_u32<<<(Nn * HH + 255) / 256, 256, 0, st>>>(emaxu, Nn * HH);
  k_zero_u32<<<(Nn * HH + 255) / 256, 256, 0, st>>>((unsigned*)denom, Nn * HH);
  k_attn_dots<<<(Nn * HH + 255) / 256, 256, 0, st>>>(hg, as, ad, asrc, adst, Nn);
  k_edge_logits<<<(E2 + 255) / 256, 256, 0, st>>>(s, d, m, EDG, Nn, asrc, adst, ebuf, emaxu);
  k_edge_w<<<(E2 + 255) / 256, 256, 0, st>>>(s, d, m, EDG, Nn, ebuf, emaxu, denom);
  k_edge_agg<<<E2, 256, 0, st>>>(s, d, m, EDG, Nn, ebuf, denom, hg, agg);
}
static void run_pool(const float* hgat, int n_per, int k,
                     const int* s_in, const int* d_in, const int* m_in,
                     int* s_out, int* d_out, int* m_out,
                     float* score, int* perm, int* newid,
                     const float* pw, const float* invn, float* xpool, hipStream_t st) {
  int Nn = BGR * n_per;
  k_scores<<<Nn, 256, 0, st>>>(hgat, pw, invn, score);
  k_topk<<<BGR, 256, 0, st>>>(score, n_per, k, perm, newid);
  k_gather<<<BGR * k, 256, 0, st>>>(hgat, score, perm, xpool);
  k_remap<<<(EDG + 255) / 256, 256, 0, st>>>(s_in, d_in, m_in, newid, s_out, d_out, m_out, EDG);
}

extern "C" void kernel_launch(void* const* d_in, const int* in_sizes, int n_in,
                              void* d_out, int out_size, void* d_ws, size_t ws_size,
                              hipStream_t stream) {
  (void)in_sizes; (void)n_in; (void)out_size; (void)ws_size;
  const float* x    = (const float*)d_in[0];
  const int*   es0  = (const int*)d_in[1];
  const int*   ed0  = (const int*)d_in[2];
  const float* W1   = (const float*)d_in[4];
  const float* as1  = (const float*)d_in[5];
  const float* ad1  = (const float*)d_in[6];
  const float* b1   = (const float*)d_in[7];
  const float* W2   = (const float*)d_in[8];
  const float* as2  = (const float*)d_in[9];
  const float* ad2  = (const float*)d_in[10];
  const float* b2   = (const float*)d_in[11];
  const float* W3   = (const float*)d_in[12];
  const float* as3  = (const float*)d_in[13];
  const float* ad3  = (const float*)d_in[14];
  const float* b3   = (const float*)d_in[15];
  const float* pw   = (const float*)d_in[16];
  const float* iW   = (const float*)d_in[17];
  const float* ib   = (const float*)d_in[18];
  const float* l1W  = (const float*)d_in[19];
  const float* l1b  = (const float*)d_in[20];
  const float* l2W  = (const float*)d_in[21];
  const float* l2b  = (const float*)d_in[22];
  float* out = (float*)d_out;

  // ---- workspace carve (bump allocator, 256B aligned) ----
  char* wsp = (char*)d_ws;
  auto alloc = [&](size_t bytes) -> void* {
    void* p = (void*)wsp;
    wsp += (bytes + 255) & ~(size_t)255;
    return p;
  };
  float*    hg    = (float*)alloc((size_t)N0 * HC * 4);      // GEMM out / xpool alias
  float*    agg   = (float*)alloc((size_t)N0 * HC * 4);      // GAT output
  float*    hiW   = (float*)alloc((size_t)BGR * KP1 * CC * 4);
  float*    asrc  = (float*)alloc((size_t)N0 * HH * 4);
  float*    adst  = (float*)alloc((size_t)N0 * HH * 4);
  unsigned* emaxu = (unsigned*)alloc((size_t)N0 * HH * 4);
  float*    denom = (float*)alloc((size_t)N0 * HH * 4);
  float*    ebuf  = (float*)alloc((size_t)(EDG + N0) * HH * 4);
  float*    score = (float*)alloc((size_t)N0 * 4);
  int*      perm  = (int*)alloc((size_t)BGR * KP1 * 4);
  int*      newid = (int*)alloc((size_t)N0 * 4);
  int*      sA    = (int*)alloc((size_t)EDG * 4);
  int*      dA    = (int*)alloc((size_t)EDG * 4);
  int*      mA    = (int*)alloc((size_t)EDG * 4);
  int*      sB    = (int*)alloc((size_t)EDG * 4);
  int*      dB    = (int*)alloc((size_t)EDG * 4);
  int*      mB    = (int*)alloc((size_t)EDG * 4);
  float*    zsum  = (float*)alloc((size_t)BGR * 512 * 4);
  float*    z1    = (float*)alloc((size_t)BGR * 1024 * 4);
  float*    invn  = (float*)alloc(256);
  float*    xpool = hg;   // xpool's live range doesn't overlap hg's

  const int N1 = BGR * KP1, N2 = BGR * KP2, N3 = BGR * KP3;

  k_zero_u32<<<(BGR * 512 + 255) / 256, 256, 0, stream>>>((unsigned*)zsum, BGR * 512);
  k_pw_invnorm<<<1, 256, 0, stream>>>(pw, invn);

  // ----- level 1 -----
  gemm(x, W1, nullptr, hg, N0, HC, 128, 0, stream);
  run_attention(hg, as1, ad1, b1, es0, ed0, nullptr, N0,
                agg, asrc, adst, emaxu, denom, ebuf, stream);
  run_pool(agg, NPG0, KP1, es0, ed0, nullptr, sA, dA, mA,
           score, perm, newid, pw, invn, xpool, stream);
  gemm(xpool, iW, ib, hiW, N1, CC, HC, 0, stream);
  k_readout<<<BGR, 256, 0, stream>>>(hiW, KP1, zsum);

  // ----- level 2 -----
  gemm(hiW, W2, nullptr, hg, N1, HC, CC, 0, stream);
  run_attention(hg, as2, ad2, b2, sA, dA, mA, N1,
                agg, asrc, adst, emaxu, denom, ebuf, stream);
  run_pool(agg, KP1, KP2, sA, dA, mA, sB, dB, mB,
           score, perm, newid, pw, invn, xpool, stream);
  gemm(xpool, iW, ib, hiW, N2, CC, HC, 0, stream);
  k_readout<<<BGR, 256, 0, stream>>>(hiW, KP2, zsum);

  // ----- level 3 -----
  gemm(hiW, W3, nullptr, hg, N2, HC, CC, 0, stream);
  run_attention(hg, as3, ad3, b3, sB, dB, mB, N2,
                agg, asrc, adst, emaxu, denom, ebuf, stream);
  run_pool(agg, KP2, KP3, sB, dB, mB, sA, dA, mA,
           score, perm, newid, pw, invn, xpool, stream);
  gemm(xpool, iW, ib, hiW, N3, CC, HC, 0, stream);
  k_readout<<<BGR, 256, 0, stream>>>(hiW, KP3, zsum);

  // ----- final MLP -----
  gemm(zsum, l1W, l1b, z1, BGR, 1024, 512, 1, stream);   // leaky_relu(0.01) fused
  k_mlp2<<<1, 64, 0, stream>>>(z1, l2W, l2b, out);
}
